// ApproxLlamaAttention_43602507989097
// MI455X (gfx1250) — matrix operand order repaired and checked by reading, not compiled, not run on hardware
//
#include <hip/hip_runtime.h>
#include <hip/hip_bf16.h>
#include <math.h>

// ---------------------------------------------------------------------------
// Llama attention block on gfx1250 (MI455X), bf16 WMMA everywhere.
// Data movement: GLOBAL_LOAD_ASYNC_TO_LDS_B128 (ASYNCcnt DMA) with double-
// buffered LDS tiles; all operands pre-transposed so staging is straight copy.
// ---------------------------------------------------------------------------

typedef __bf16 bf16;
typedef __attribute__((ext_vector_type(16))) __bf16 v16bf;
typedef __attribute__((ext_vector_type(8)))  __bf16 bf16x8;
typedef __attribute__((ext_vector_type(4)))  __bf16 bf16x4;
typedef __attribute__((ext_vector_type(8)))  float  v8f;

#define DEV_INLINE __device__ __forceinline__

constexpr int S_   = 2048;
constexpr int D_   = 4096;
constexpr int H_   = 32;
constexpr int HKV_ = 8;
constexpr int HD_  = 128;

DEV_INLINE v16bf cat8(bf16x8 lo, bf16x8 hi) {
  return __builtin_shufflevector(lo, hi, 0,1,2,3,4,5,6,7,8,9,10,11,12,13,14,15);
}

DEV_INLINE v8f wmma_bf16(v16bf a, v16bf b, v8f c) {
  // D = A(16x32 bf16) * B(32x16 bf16) + C(16x16 f32)
  return __builtin_amdgcn_wmma_f32_16x16x32_bf16(
      /*neg_a=*/false, a, /*neg_b=*/false, b,
      /*c_mod=*/(short)0, c, /*reuse_a=*/false, /*reuse_b=*/false);
}

// Async DMA: copy 16B per lane from global memory directly into LDS.
// Tracked with ASYNCcnt; completion enforced via s_wait_asynccnt.
DEV_INLINE void async_ld_b128(void* lds_ptr, const void* gptr) {
  unsigned lds_off = (unsigned)(unsigned long long)lds_ptr;  // low 32b = LDS offset
  asm volatile("global_load_async_to_lds_b128 %0, %1, off"
               :: "v"(lds_off), "v"(gptr) : "memory");
}
#define ASYNC_WAIT(N) asm volatile("s_wait_asynccnt " #N ::: "memory")

// ---------------------------------------------------------------------------
// fp32 -> bf16 elementwise conversion (sizes are multiples of 4)
// ---------------------------------------------------------------------------
__global__ void cvt_f32_bf16(const float* __restrict__ in,
                             bf16* __restrict__ out, int n) {
  int i = (blockIdx.x * blockDim.x + threadIdx.x) * 4;
  if (i >= n) return;
  float4 f = *(const float4*)(in + i);
  bf16x4 o;
  o[0] = (bf16)f.x; o[1] = (bf16)f.y; o[2] = (bf16)f.z; o[3] = (bf16)f.w;
  *(bf16x4*)(out + i) = o;
}

// ---------------------------------------------------------------------------
// Tiled transpose + fp32->bf16: out[c][r] = (bf16) in[r][c].  in: R x C fp32.
// Grid (C/32, R/32), block 256 (32 x 8).
// ---------------------------------------------------------------------------
__global__ __launch_bounds__(256)
void transpose_f32_bf16(const float* __restrict__ in, bf16* __restrict__ out,
                        int R, int C) {
  __shared__ float tile[32][33];
  const int r0 = blockIdx.y * 32, c0 = blockIdx.x * 32;
  const int tx = threadIdx.x & 31, ty = threadIdx.x >> 5;
  #pragma unroll
  for (int i = ty; i < 32; i += 8)
    tile[i][tx] = in[(size_t)(r0 + i) * C + c0 + tx];
  __syncthreads();
  #pragma unroll
  for (int i = ty; i < 32; i += 8)
    out[(size_t)(c0 + i) * R + r0 + tx] = (bf16)tile[tx][i];
}

// ---------------------------------------------------------------------------
// C(MxN fp32) = A(MxK bf16, row-major) * Bt(NxK bf16, row-major = B^T)
// 128x128 tile per 256-thread block (8 waves, 4x2 wave grid, 2x4 16x16 tiles
// per wave), K-step 32.  Tiles staged via async DMA, double-buffered.
// ---------------------------------------------------------------------------
__global__ __launch_bounds__(256)
void gemm_bf16_wmma(const bf16* __restrict__ A, const bf16* __restrict__ Bt,
                    float* __restrict__ C, int M, int N, int K) {
  __shared__ bf16 As[2][128][40];   // [m][k], pad 32->40
  __shared__ bf16 Bs[2][128][40];   // [n][k], pad 32->40

  const int tid  = threadIdx.x;
  const int lane = tid & 31;
  const int wid  = tid >> 5;
  const int wm   = wid >> 1;          // 0..3 : 32-row strip
  const int wn   = wid & 1;           // 0..1 : 64-col strip
  const int bm   = blockIdx.y * 128;
  const int bn   = blockIdx.x * 128;
  const int khalf = lane >> 4;
  const int ln    = lane & 15;

  v8f acc[2][4];
  #pragma unroll
  for (int i = 0; i < 2; ++i)
    #pragma unroll
    for (int j = 0; j < 4; ++j)
      #pragma unroll
      for (int e = 0; e < 8; ++e) acc[i][j][e] = 0.0f;

  // Per tile: A 128x32 + B 128x32 = 2x512 chunks of 8 bf16; 2+2 async/thread.
  auto issue = [&](int kt, int buf) {
    int k0 = kt * 32;
    #pragma unroll
    for (int c0 = 0; c0 < 2; ++c0) {
      int c = tid + c0 * 256;
      int row = c >> 2;
      int kc  = (c & 3) * 8;
      async_ld_b128(&As[buf][row][kc], A  + (size_t)(bm + row) * K + k0 + kc);
      async_ld_b128(&Bs[buf][row][kc], Bt + (size_t)(bn + row) * K + k0 + kc);
    }
  };

  const int nk = K >> 5;
  issue(0, 0);
  for (int kt = 0; kt < nk; ++kt) {
    const int buf = kt & 1;
    if (kt + 1 < nk) { issue(kt + 1, buf ^ 1); ASYNC_WAIT(4); }
    else             { ASYNC_WAIT(0); }
    __syncthreads();                       // tile `kt` visible to all waves

    // A fragments (ISA 7.12.2): lane<16 K{0..7,16..23}, lane>=16 K{8..15,24..31}
    v16bf af[2];
    #pragma unroll
    for (int tm = 0; tm < 2; ++tm) {
      int row = wm * 32 + tm * 16 + ln;
      bf16x8 lo = *(const bf16x8*)&As[buf][row][khalf * 8];
      bf16x8 hi = *(const bf16x8*)&As[buf][row][16 + khalf * 8];
      af[tm] = cat8(lo, hi);
    }
    // B fragments: lane<16 K{0..7,16..23} of col n, lane>=16 K{8..15,24..31}
    v16bf bfr[4];
    #pragma unroll
    for (int tn = 0; tn < 4; ++tn) {
      int col = wn * 64 + tn * 16 + ln;
      bf16x8 lo = *(const bf16x8*)&Bs[buf][col][khalf * 8];
      bf16x8 hi = *(const bf16x8*)&Bs[buf][col][16 + khalf * 8];
      bfr[tn] = cat8(lo, hi);
    }
    #pragma unroll
    for (int tm = 0; tm < 2; ++tm)
      #pragma unroll
      for (int tn = 0; tn < 4; ++tn)
        acc[tm][tn] = wmma_bf16(af[tm], bfr[tn], acc[tm][tn]);

    __syncthreads();                       // all reads done before buffer reuse
  }

  const int mhalf = khalf * 8;
  #pragma unroll
  for (int tm = 0; tm < 2; ++tm)
    #pragma unroll
    for (int tn = 0; tn < 4; ++tn)
      #pragma unroll
      for (int r = 0; r < 8; ++r) {
        int row = bm + wm * 32 + tm * 16 + mhalf + r;
        int col = bn + wn * 64 + tn * 16 + ln;
        C[(size_t)row * N + col] = acc[tm][tn][r];
      }
}

// ---------------------------------------------------------------------------
// RoPE: read fp32 proj, rotate pairs (d, d+64), emit bf16 for WMMA attention.
// ---------------------------------------------------------------------------
__global__ void rope_f32_to_bf16(const float* __restrict__ in,
                                 bf16* __restrict__ out,
                                 const int* __restrict__ pos_ids, int nh) {
  int idx = blockIdx.x * blockDim.x + threadIdx.x;
  int total = S_ * nh * (HD_ / 2);
  if (idx >= total) return;
  int d  = idx & 63;
  int t  = idx >> 6;
  int hh = t % nh;
  int s  = t / nh;
  size_t base = ((size_t)s * nh + hh) * HD_;
  float x1 = in[base + d];
  float x2 = in[base + d + 64];
  float pos  = (float)pos_ids[s];
  float invf = __powf(10000.0f, -(float)(2 * d) * (1.0f / 128.0f));
  float fr = pos * invf;
  float c, sn;
  __sincosf(fr, &sn, &c);
  out[base + d]      = (bf16)(x1 * c - x2 * sn);
  out[base + d + 64] = (bf16)(x2 * c + x1 * sn);
}

// ---------------------------------------------------------------------------
// Flash attention, causal, GQA (4 Q heads per KV head).
// Block = 4 waves x 32.  blockIdx = (q_block of 64 rows, head).
// Each wave owns 16 query rows.  K tiles ([kk][d] row-major) and pre-
// transposed V tiles ([d][kk]) staged via async DMA, double-buffered.
// ---------------------------------------------------------------------------
__global__ __launch_bounds__(128)
void flash_attn_wmma(const bf16* __restrict__ Q, const bf16* __restrict__ Kc,
                     const bf16* __restrict__ Vtg, float* __restrict__ O) {
  __shared__ bf16 Ks[2][32][HD_ + 8];  // K tile row-major (B colmajor for QK^T)
  __shared__ bf16 Vt[2][HD_][40];      // V^T tile (B colmajor for PV)
  __shared__ bf16 Ps[4][16][40];       // per-wave P staging (C -> A layout)

  const int tid  = threadIdx.x;
  const int lane = tid & 31;
  const int wid  = tid >> 5;
  const int h    = blockIdx.y;
  const int hkv  = h >> 2;                      // H/HKV = 4
  const int q0   = blockIdx.x * 64 + wid * 16;
  const int khalf = lane >> 4;
  const int ln    = lane & 15;

  // Q fragments: 16 rows x 128 dims = 4 A-frags of 16x32
  v16bf qf[4];
  {
    const bf16* qrow = Q + ((size_t)(q0 + ln) * H_ + h) * HD_;
    #pragma unroll
    for (int c = 0; c < 4; ++c) {
      bf16x8 lo = *(const bf16x8*)(qrow + c * 32 + khalf * 8);
      bf16x8 hi = *(const bf16x8*)(qrow + c * 32 + 16 + khalf * 8);
      qf[c] = cat8(lo, hi);
    }
  }

  v8f o[8];
  #pragma unroll
  for (int t = 0; t < 8; ++t)
    #pragma unroll
    for (int e = 0; e < 8; ++e) o[t][e] = 0.0f;
  float m8[8], l8[8];
  #pragma unroll
  for (int r = 0; r < 8; ++r) { m8[r] = -1e30f; l8[r] = 0.0f; }

  // Per tile: K 32x128 + V^T 128x32 = 2x512 chunks of 8; 4+4 async per thread.
  auto issue = [&](int kt, int buf) {
    #pragma unroll
    for (int c0 = 0; c0 < 4; ++c0) {
      int c  = tid + c0 * 128;
      int kk = c >> 4;
      int dc = (c & 15) * 8;
      async_ld_b128(&Ks[buf][kk][dc],
                    Kc + ((size_t)(kt * 32 + kk) * HKV_ + hkv) * HD_ + dc);
    }
    #pragma unroll
    for (int c0 = 0; c0 < 4; ++c0) {
      int c  = tid + c0 * 128;
      int d  = c >> 2;
      int kc = (c & 3) * 8;
      async_ld_b128(&Vt[buf][d][kc],
                    Vtg + (size_t)(hkv * HD_ + d) * S_ + kt * 32 + kc);
    }
  };

  const int nkt = blockIdx.x * 2 + 2;           // causal: keys up to block end
  const float scale = 0.08838834764831845f;     // 1/sqrt(128)

  issue(0, 0);
  for (int kt = 0; kt < nkt; ++kt) {
    const int buf = kt & 1;
    if (kt + 1 < nkt) { issue(kt + 1, buf ^ 1); ASYNC_WAIT(8); }
    else              { ASYNC_WAIT(0); }
    __syncthreads();

    const bool active = (kt * 32 <= q0 + 15);
    if (active) {
      // S = Q K^T : 16x32 scores = two 16x16 n-tiles, 4 WMMAs each over d
      v8f sacc[2];
      #pragma unroll
      for (int t = 0; t < 2; ++t)
        #pragma unroll
        for (int e = 0; e < 8; ++e) sacc[t][e] = 0.0f;
      #pragma unroll
      for (int t = 0; t < 2; ++t) {
        int kkcol = t * 16 + ln;
        #pragma unroll
        for (int c = 0; c < 4; ++c) {
          bf16x8 lo = *(const bf16x8*)&Ks[buf][kkcol][c * 32 + khalf * 8];
          bf16x8 hi = *(const bf16x8*)&Ks[buf][kkcol][c * 32 + 16 + khalf * 8];
          sacc[t] = wmma_bf16(qf[c], cat8(lo, hi), sacc[t]);
        }
      }
      // Online softmax (rows live in C-layout VGPR index; reduce across the
      // 16-lane N-group with wave32 shuffles)
      #pragma unroll
      for (int r = 0; r < 8; ++r) {
        int row = q0 + khalf * 8 + r;
        float s0 = sacc[0][r] * scale;
        float s1 = sacc[1][r] * scale;
        if (kt * 32 + ln > row)      s0 = -1e30f;
        if (kt * 32 + 16 + ln > row) s1 = -1e30f;
        float pm = fmaxf(s0, s1);
        #pragma unroll
        for (int off = 8; off >= 1; off >>= 1)
          pm = fmaxf(pm, __shfl_xor(pm, off, 32));
        float mnew = fmaxf(m8[r], pm);
        float p0 = __expf(s0 - mnew);
        float p1 = __expf(s1 - mnew);
        float psum = p0 + p1;
        #pragma unroll
        for (int off = 8; off >= 1; off >>= 1)
          psum += __shfl_xor(psum, off, 32);
        float alpha = __expf(m8[r] - mnew);
        l8[r] = l8[r] * alpha + psum;
        m8[r] = mnew;
        #pragma unroll
        for (int t = 0; t < 8; ++t) o[t][r] *= alpha;
        Ps[wid][khalf * 8 + r][ln]      = (bf16)p0;
        Ps[wid][khalf * 8 + r][16 + ln] = (bf16)p1;
      }
    }
    __syncthreads();   // Ps C->A layout round trip (cross-lane via LDS)
    if (active) {
      bf16x8 plo = *(const bf16x8*)&Ps[wid][ln][khalf * 8];
      bf16x8 phi = *(const bf16x8*)&Ps[wid][ln][16 + khalf * 8];
      v16bf pf = cat8(plo, phi);
      #pragma unroll
      for (int t = 0; t < 8; ++t) {
        int dcol = t * 16 + ln;
        bf16x8 lo = *(const bf16x8*)&Vt[buf][dcol][khalf * 8];
        bf16x8 hi = *(const bf16x8*)&Vt[buf][dcol][16 + khalf * 8];
        o[t] = wmma_bf16(pf, cat8(lo, hi), o[t]);
      }
    }
    __syncthreads();   // all reads done before async overwrites this buffer
  }

  #pragma unroll
  for (int r = 0; r < 8; ++r) {
    float inv = (l8[r] > 0.0f) ? 1.0f / l8[r] : 0.0f;
    int row = q0 + khalf * 8 + r;
    float* orow = O + ((size_t)row * H_ + h) * HD_;
    #pragma unroll
    for (int t = 0; t < 8; ++t) orow[t * 16 + ln] = o[t][r] * inv;
  }
}

// ---------------------------------------------------------------------------
extern "C" void kernel_launch(void* const* d_in, const int* in_sizes, int n_in,
                              void* d_out, int out_size, void* d_ws,
                              size_t ws_size, hipStream_t stream) {
  (void)in_sizes; (void)n_in; (void)out_size; (void)ws_size;
  const float* X   = (const float*)d_in[0];
  const float* Wq  = (const float*)d_in[1];
  const float* Wk  = (const float*)d_in[2];
  const float* Wv  = (const float*)d_in[3];
  const float* Wo  = (const float*)d_in[4];
  const int*   pos = (const int*)d_in[5];
  float* out = (float*)d_out;

  size_t off = 0;
  auto alloc = [&](size_t bytes) -> char* {
    char* p = (char*)d_ws + off;
    off += (bytes + 255) & ~(size_t)255;
    return p;
  };
  bf16*  Xb   = (bf16*)alloc((size_t)S_ * D_ * 2);
  bf16*  Wqt  = (bf16*)alloc((size_t)D_ * H_ * HD_ * 2);   // [n][k]
  bf16*  Wkt  = (bf16*)alloc((size_t)D_ * HKV_ * HD_ * 2); // [n][k]
  bf16*  Wvt  = (bf16*)alloc((size_t)D_ * HKV_ * HD_ * 2); // [n][k]
  bf16*  Wot  = (bf16*)alloc((size_t)H_ * HD_ * D_ * 2);   // [n][k]
  float* Qf   = (float*)alloc((size_t)S_ * H_ * HD_ * 4);
  float* Kf   = (float*)alloc((size_t)S_ * HKV_ * HD_ * 4);
  float* Vf   = (float*)alloc((size_t)S_ * HKV_ * HD_ * 4);
  bf16*  Qb   = (bf16*)alloc((size_t)S_ * H_ * HD_ * 2);
  bf16*  Kb   = (bf16*)alloc((size_t)S_ * HKV_ * HD_ * 2);
  bf16*  Vtb  = (bf16*)alloc((size_t)S_ * HKV_ * HD_ * 2); // [hkv*HD][s]
  float* Af   = (float*)alloc((size_t)S_ * H_ * HD_ * 4);
  bf16*  Ab   = (bf16*)alloc((size_t)S_ * H_ * HD_ * 2);

  auto cvt = [&](const float* src, bf16* dst, size_t n) {
    cvt_f32_bf16<<<(unsigned)(n / 1024), 256, 0, stream>>>(src, dst, (int)n);
  };
  auto tr = [&](const float* src, bf16* dst, int R, int C) {
    transpose_f32_bf16<<<dim3(C / 32, R / 32), 256, 0, stream>>>(src, dst, R, C);
  };

  // 1) activations -> bf16; weights -> transposed bf16 ([n][k])
  cvt(X, Xb, (size_t)S_ * D_);
  tr(Wq, Wqt, D_, H_ * HD_);
  tr(Wk, Wkt, D_, HKV_ * HD_);
  tr(Wv, Wvt, D_, HKV_ * HD_);
  tr(Wo, Wot, H_ * HD_, D_);

  // 2) QKV projections (K-dim 4096)
  gemm_bf16_wmma<<<dim3((H_ * HD_) / 128, S_ / 128), 256, 0, stream>>>(
      Xb, Wqt, Qf, S_, H_ * HD_, D_);
  gemm_bf16_wmma<<<dim3((HKV_ * HD_) / 128, S_ / 128), 256, 0, stream>>>(
      Xb, Wkt, Kf, S_, HKV_ * HD_, D_);
  gemm_bf16_wmma<<<dim3((HKV_ * HD_) / 128, S_ / 128), 256, 0, stream>>>(
      Xb, Wvt, Vf, S_, HKV_ * HD_, D_);

  // 3) RoPE (Q, K) -> bf16; V -> transposed bf16 [hkv*HD][s]
  rope_f32_to_bf16<<<(S_ * H_ * 64) / 256, 256, 0, stream>>>(Qf, Qb, pos, H_);
  rope_f32_to_bf16<<<(S_ * HKV_ * 64) / 256, 256, 0, stream>>>(Kf, Kb, pos, HKV_);
  tr(Vf, Vtb, S_, HKV_ * HD_);

  // 4) causal flash attention
  flash_attn_wmma<<<dim3(S_ / 64, H_), 128, 0, stream>>>(Qb, Kb, Vtb, Af);

  // 5) output projection
  cvt(Af, Ab, (size_t)S_ * H_ * HD_);
  gemm_bf16_wmma<<<dim3(D_ / 128, S_ / 128), 256, 0, stream>>>(
      Ab, Wot, out, S_, D_, H_ * HD_);
}
